// JTNNDecoder_27934467293755
// MI455X (gfx1250) — matrix operand order repaired and checked by reading, not compiled, not run on hardware
//
#include <hip/hip_runtime.h>
#include <hip/hip_bf16.h>
#include <stdint.h>

// ---------------- problem constants ----------------
#define B_  1024           // batch (trees)
#define NN  16             // nodes per tree
#define EPT 30             // directed edges per tree
#define H_  450            // hidden
#define HP  480            // hidden padded for K (x32)
#define HO  464            // hidden padded for N (x16)
#define L_  56             // tree_vec len
#define LP  64             // padded
#define V_  780            // vocab
#define VP  784            // padded

typedef __attribute__((ext_vector_type(16))) __bf16 v16bf;
typedef __attribute__((ext_vector_type(8)))  float  v8f;

// ---------------- static tree tables (device) ----------------
__constant__ int   c_src[30]      = {0,1,0,2,1,3,1,4,2,5,2,6,3,7,3,8,4,9,4,10,5,11,5,12,6,13,6,14,7,15};
__constant__ int   c_pred_dst[16] = {0,1,3,7,15,8,4,9,10,2,5,11,12,6,13,14};
__constant__ float c_stop_t[31]   = {1,1,1,1,0,0,1,0,0,1,1,0,1,0,0,0,1,1,1,0,1,0,0,1,1,0,1,0,0,0,0};

__device__ __forceinline__ unsigned short f2bf(float f) {
  unsigned int u = __float_as_uint(f);
  u += 0x7FFFu + ((u >> 16) & 1u);          // round-to-nearest-even
  return (unsigned short)(u >> 16);
}

// ---------------- generic bf16 WMMA GEMM: C = act(A @ W^T + bias + addend) ----------------
// A: bf16 [M, lda]   (amode=1: row r gathers from x_bf row (r/30)*16 + src[r%30])
// W: bf16 [Ntiles*16, ldw] row-major (row = output col), zero-padded
// outputs: fp32 C [M, ldc] and/or bf16 Cbf [M, ldcbf]; pad cols written as 0
__global__ __launch_bounds__(128) void gemm_bf16_wmma(
    const unsigned short* __restrict__ A, int lda, int amode,
    const unsigned short* __restrict__ W, int ldw,
    const float* __restrict__ bias,
    const float* __restrict__ addend, int ld_add, int add_e0,
    float* __restrict__ C, int ldc,
    unsigned short* __restrict__ Cbf, int ldcbf,
    int Ncols, int Ntiles, int Kpad, int act)
{
  const int lane  = threadIdx.x & 31;
  const int wv    = threadIdx.x >> 5;
  const int tileN = blockIdx.y * 4 + wv;
  if (tileN >= Ntiles) return;               // uniform per wave -> EXEC all-ones inside
  const int m    = lane & 15;
  const int half = lane >> 4;
  const int rowA = blockIdx.x * 16 + m;
  int ar = rowA;
  if (amode == 1) { int b = rowA / 30; int e = rowA - b * 30; ar = b * 16 + c_src[e]; }
  const unsigned short* Ap = A + (size_t)ar * lda;
  const unsigned short* Wp = W + (size_t)(tileN * 16 + m) * ldw;   // lane m == column n
  v8f acc = {};
  for (int k0 = 0; k0 < Kpad; k0 += 32) {
    union { v16bf v; uint4 u[2]; } av, bv;
    // A 16x32 bf16 layout: half0 lanes need K[k0+0..7],[k0+16..23]; half1: [k0+8..15],[k0+24..31]
    const uint4* ap = reinterpret_cast<const uint4*>(Ap + k0 + 8 * half);
    av.u[0] = ap[0];
    av.u[1] = ap[2];
    // B 32x16 bf16 layout: same K order as A: half0 K[k0+0..7],[k0+16..23]; half1: [k0+8..15],[k0+24..31]
    const uint4* bp = reinterpret_cast<const uint4*>(Wp + k0 + 8 * half);
    bv.u[0] = bp[0];
    bv.u[1] = bp[2];
    acc = __builtin_amdgcn_wmma_f32_16x16x32_bf16(false, av.v, false, bv.v,
                                                  (short)0, acc, false, false);
  }
  const int col = tileN * 16 + m;
  #pragma unroll
  for (int v = 0; v < 8; ++v) {
    int row = blockIdx.x * 16 + v + 8 * half;   // C/D layout: VGPR v holds M = v + 8*half
    float val = 0.f;
    if (col < Ncols) {
      val = acc[v];
      if (bias) val += bias[col];
      if (addend) {
        int arow = (add_e0 >= 0) ? ((row & (B_ - 1)) * 30 + add_e0) : row;
        val += addend[(size_t)arow * ld_add + col];
      }
      if (act == 1)      val = 1.f / (1.f + __expf(-val));
      else if (act == 2) val = tanhf(val);
      else if (act == 3) val = fmaxf(val, 0.f);
    }
    if (C)   C[(size_t)row * ldc + col] = val;
    if (Cbf) Cbf[(size_t)row * ldcbf + col] = f2bf(val);
  }
}

// ---------------- elementwise kernels ----------------
__global__ void k_pack(const float* __restrict__ src, int src_ld, int s_col0,
                       unsigned short* __restrict__ dst, int dst_ld, int d_col0,
                       int rows, int cols) {
  int i = blockIdx.x * blockDim.x + threadIdx.x;
  if (i >= rows * cols) return;
  int r = i / cols, c = i - r * cols;
  dst[(size_t)r * dst_ld + d_col0 + c] = f2bf(src[(size_t)r * src_ld + s_col0 + c]);
}

__global__ void k_embed(const int* __restrict__ wid, const float* __restrict__ emb,
                        unsigned short* __restrict__ x_bf) {
  int i = blockIdx.x * blockDim.x + threadIdx.x;
  if (i >= B_ * NN * HP) return;
  int r = i / HP, c = i - r * HP;
  float v = (c < H_) ? emb[(size_t)wid[r] * H_ + c] : 0.f;
  x_bf[i] = f2bf(v);
}

__global__ void k_tv(const float* __restrict__ tree, unsigned short* __restrict__ tv_bf) {
  int i = blockIdx.x * blockDim.x + threadIdx.x;
  if (i >= B_ * LP) return;
  int b = i / LP, c = i - b * LP;
  tv_bf[i] = f2bf(c < L_ ? tree[(size_t)b * L_ + c] : 0.f);
}

__global__ void k_tgt(const int* __restrict__ wid, int* __restrict__ tgt) {
  int i = blockIdx.x * blockDim.x + threadIdx.x;
  if (i >= 16 * B_) return;
  int k = i >> 10, b = i & (B_ - 1);
  tgt[i] = wid[b * NN + c_pred_dst[k]];
}

// gather h_nei (bf16) + sum_h (fp32 and bf16 into A_z), copy sx into A_z/A_h
__global__ void k_step_gather(const float* __restrict__ h, const unsigned short* __restrict__ x_bf,
                              unsigned short* __restrict__ hnei, float* __restrict__ sum_h,
                              unsigned short* __restrict__ Az, unsigned short* __restrict__ Ah,
                              int e0, int d, int p0, int p1) {
  int i = blockIdx.x * blockDim.x + threadIdx.x;
  if (i >= B_ * HP) return;
  int b = i / HP, c = i - b * HP;
  unsigned short sx = x_bf[(size_t)(b * NN + c_src[e0]) * HP + c];
  Az[(size_t)b * 960 + c] = sx;
  Ah[(size_t)b * 960 + c] = sx;
  float s = 0.f;
  if (d > 0) {
    if (c < H_) {
      float h0 = h[(size_t)(b * EPT + p0) * HO + c];
      hnei[(size_t)b * HP + c] = f2bf(h0);
      s = h0;
      if (d > 1) {
        float h1 = h[(size_t)(b * EPT + p1) * HO + c];
        hnei[(size_t)(B_ + b) * HP + c] = f2bf(h1);
        s += h1;
      }
    } else {
      hnei[(size_t)b * HP + c] = 0;
      if (d > 1) hnei[(size_t)(B_ + b) * HP + c] = 0;
    }
  }
  if (c < HO) sum_h[(size_t)b * HO + c] = s;
  Az[(size_t)b * 960 + 480 + c] = f2bf(s);
  if (d == 0) Ah[(size_t)b * 960 + 480 + c] = 0;
}

// sum_g = sum_pi r[pi]*h_nei[pi]  (bf16 into A_h second block)
__global__ void k_sumg(const float* __restrict__ h, const float* __restrict__ r_buf,
                       unsigned short* __restrict__ Ah, int d, int p0, int p1) {
  int i = blockIdx.x * blockDim.x + threadIdx.x;
  if (i >= B_ * HP) return;
  int b = i / HP, c = i - b * HP;
  float s = 0.f;
  if (c < H_) {
    s = r_buf[(size_t)b * HO + c] * h[(size_t)(b * EPT + p0) * HO + c];
    if (d > 1) s += r_buf[(size_t)(B_ + b) * HO + c] * h[(size_t)(b * EPT + p1) * HO + c];
  }
  Ah[(size_t)b * 960 + 480 + c] = f2bf(s);
}

// new_h update + stop-row (t) + optional pred-row (predk)
__global__ void k_step_final(const float* __restrict__ z, const float* __restrict__ pre,
                             const float* __restrict__ sum_h, float* __restrict__ h,
                             const unsigned short* __restrict__ x_bf,
                             const unsigned short* __restrict__ tv_bf,
                             unsigned short* __restrict__ stop_buf,
                             unsigned short* __restrict__ pred_buf,
                             int e0, int t, int predk) {
  int i = blockIdx.x * blockDim.x + threadIdx.x;
  if (i >= B_ * 1024) return;
  int b = i >> 10, cc = i & 1023;
  size_t srow = ((size_t)t * B_ + b) * 1024;
  if (cc < 480) {
    stop_buf[srow + cc] = x_bf[(size_t)(b * NN + c_src[e0]) * HP + cc];
  } else if (cc < 960) {
    int c = cc - 480;
    float curo = 0.f;
    if (c < H_) {
      float sh = sum_h[(size_t)b * HO + c];
      curo = sh + h[(size_t)(b * EPT + (e0 ^ 1)) * HO + c];   // reverse edge (0 if unvisited)
      float zz = z[(size_t)b * HO + c];
      float nh = (1.f - zz) * sh + zz * pre[(size_t)b * HO + c];
      h[(size_t)(b * EPT + e0) * HO + c] = nh;
      if (predk >= 0) pred_buf[((size_t)predk * B_ + b) * 544 + c] = f2bf(nh);
    } else {
      if (c < HO) h[(size_t)(b * EPT + e0) * HO + c] = 0.f;
      if (predk >= 0) pred_buf[((size_t)predk * B_ + b) * 544 + c] = 0;
    }
    stop_buf[srow + cc] = f2bf(curo);
  } else {
    int c = cc - 960;
    unsigned short tvv = tv_bf[(size_t)b * LP + c];
    stop_buf[srow + cc] = tvv;
    if (predk >= 0) pred_buf[((size_t)predk * B_ + b) * 544 + 480 + c] = tvv;
  }
}

__global__ void k_pred_row0(const unsigned short* __restrict__ tv_bf,
                            unsigned short* __restrict__ pred_buf) {
  int i = blockIdx.x * blockDim.x + threadIdx.x;
  if (i >= B_ * 544) return;
  int b = i / 544, c = i - b * 544;
  pred_buf[(size_t)b * 544 + c] = (c >= 480) ? tv_bf[(size_t)b * LP + (c - 480)] : 0;
}

__global__ void k_root_stop(const float* __restrict__ h, const unsigned short* __restrict__ x_bf,
                            const unsigned short* __restrict__ tv_bf,
                            unsigned short* __restrict__ stop_buf) {
  int i = blockIdx.x * blockDim.x + threadIdx.x;
  if (i >= B_ * 1024) return;
  int b = i >> 10, cc = i & 1023;
  size_t srow = ((size_t)30 * B_ + b) * 1024;
  unsigned short v;
  if (cc < 480) v = x_bf[(size_t)(b * NN) * HP + cc];
  else if (cc < 960) {
    int c = cc - 480;
    float s = 0.f;
    if (c < H_) s = h[(size_t)(b * EPT + 1) * HO + c] + h[(size_t)(b * EPT + 3) * HO + c];
    v = f2bf(s);
  } else v = tv_bf[(size_t)b * LP + (cc - 960)];
  stop_buf[srow + cc] = v;
}

// softmax-CE over V_=780 cols; one wave per row, 8 rows/block; deterministic partials
__global__ __launch_bounds__(256) void k_pred_reduce(const float* __restrict__ scores,
                                                     const int* __restrict__ tgt,
                                                     float* __restrict__ partial) {
  int w = threadIdx.x >> 5, lane = threadIdx.x & 31;
  int row = blockIdx.x * 8 + w;
  const float* sr = scores + (size_t)row * VP;
  float mv = -3.4e38f; int mi = V_;
  for (int c = lane; c < V_; c += 32) {
    float v = sr[c];
    if (v > mv || (v == mv && c < mi)) { mv = v; mi = c; }
  }
  for (int o = 16; o; o >>= 1) {
    float ov = __shfl_xor(mv, o, 32);
    int   oi = __shfl_xor(mi, o, 32);
    if (ov > mv || (ov == mv && oi < mi)) { mv = ov; mi = oi; }
  }
  float se = 0.f;
  for (int c = lane; c < V_; c += 32) se += __expf(sr[c] - mv);
  for (int o = 16; o; o >>= 1) se += __shfl_xor(se, o, 32);
  __shared__ float sl[8], sa[8];
  if (lane == 0) {
    int tg = tgt[row];
    sl[w] = (logf(se) + mv) - sr[tg];
    sa[w] = (mi == tg) ? 1.f : 0.f;
  }
  __syncthreads();
  if (threadIdx.x == 0) {
    float Lx = 0, Ax = 0;
    for (int k = 0; k < 8; ++k) { Lx += sl[k]; Ax += sa[k]; }
    partial[(size_t)blockIdx.x * 2]     = Lx;
    partial[(size_t)blockIdx.x * 2 + 1] = Ax;
  }
}

__global__ __launch_bounds__(256) void k_stop_reduce(const float* __restrict__ s_buf,
                                                     float* __restrict__ partial) {
  int row = blockIdx.x * 256 + threadIdx.x;           // 31744 rows = 124 blocks exactly
  float s  = s_buf[(size_t)row * 16];
  float st = c_stop_t[row >> 10];
  float sp = fmaxf(s, 0.f) + log1pf(__expf(-fabsf(s)));
  float l  = sp - s * st;
  float a  = (((s >= 0.f) ? 1.f : 0.f) == st) ? 1.f : 0.f;
  __shared__ float sl[256], sa[256];
  sl[threadIdx.x] = l; sa[threadIdx.x] = a;
  __syncthreads();
  for (int o = 128; o; o >>= 1) {
    if (threadIdx.x < o) { sl[threadIdx.x] += sl[threadIdx.x + o]; sa[threadIdx.x] += sa[threadIdx.x + o]; }
    __syncthreads();
  }
  if (threadIdx.x == 0) {
    partial[(size_t)blockIdx.x * 2]     = sl[0];
    partial[(size_t)blockIdx.x * 2 + 1] = sa[0];
  }
}

__global__ void k_final(const float* __restrict__ pp, const float* __restrict__ sp_,
                        float* __restrict__ out) {
  if (threadIdx.x || blockIdx.x) return;
  double pl = 0, pa = 0, sl = 0, sa = 0;
  for (int i = 0; i < 2048; ++i) { pl += pp[2 * i]; pa += pp[2 * i + 1]; }
  for (int i = 0; i < 124;  ++i) { sl += sp_[2 * i]; sa += sp_[2 * i + 1]; }
  out[0] = (float)(pl / 1024.0);
  out[1] = (float)(sl / 1024.0);
  out[2] = (float)(pa / 16384.0);
  out[3] = (float)(sa / 31744.0);
}

// ---------------- host driver ----------------
extern "C" void kernel_launch(void* const* d_in, const int* in_sizes, int n_in,
                              void* d_out, int out_size, void* d_ws, size_t ws_size,
                              hipStream_t stream) {
  (void)in_sizes; (void)n_in; (void)out_size; (void)ws_size;
  const int*   wid   = (const int*)  d_in[0];
  const float* tree  = (const float*)d_in[1];
  const float* emb   = (const float*)d_in[2];
  const float* W_r   = (const float*)d_in[3];
  const float* U_r_w = (const float*)d_in[4];
  const float* U_r_b = (const float*)d_in[5];
  const float* W_z_w = (const float*)d_in[6];
  const float* W_z_b = (const float*)d_in[7];
  const float* W_h_w = (const float*)d_in[8];
  const float* W_h_b = (const float*)d_in[9];
  const float* W_w   = (const float*)d_in[10];
  const float* W_b   = (const float*)d_in[11];
  const float* U_w   = (const float*)d_in[12];
  const float* U_b   = (const float*)d_in[13];
  const float* W_o_w = (const float*)d_in[14];
  const float* W_o_b = (const float*)d_in[15];
  const float* U_s_w = (const float*)d_in[16];
  const float* U_s_b = (const float*)d_in[17];
  float* out = (float*)d_out;

  // DFS schedule and per-edge predecessor lists (host-side static tree)
  static const int SCHED_E[30] = {0,4,12,28,29,13,14,15,5,6,16,17,18,19,7,1,2,8,20,21,22,23,9,10,24,25,26,27,11,3};
  static const int SCHED_P[30] = {0,0,0,0,1,1,0,1,1,0,0,1,0,1,1,1,0,0,0,1,0,1,1,0,0,1,0,1,1,1};
  static const int EDGE_D[30]  = {1,2,1,2,2,2,2,2,2,2,2,2,2,1,2,0,2,0,2,0,2,0,2,0,2,0,2,0,1,0};
  static const int EDGE_P0[30] = {3,5,1,9,0,13,0,17,2,21,2,25,4,29,4,0,6,0,6,0,8,0,8,0,10,0,10,0,12,0};
  static const int EDGE_P1[30] = {0,7,0,11,7,15,5,19,11,23,9,27,15,0,13,0,19,0,17,0,23,0,21,0,27,0,25,0,0,0};

  // ---- workspace carve (~340 MB) ----
  char* cur = (char*)d_ws;
  auto alloc = [&](size_t bytes) -> char* {
    char* r = cur; cur += (bytes + 255) & ~(size_t)255; return r;
  };
  unsigned short* x_bf    = (unsigned short*)alloc((size_t)B_ * NN * HP * 2);
  unsigned short* tv_bf   = (unsigned short*)alloc((size_t)B_ * LP * 2);
  unsigned short* Wr_bf   = (unsigned short*)alloc((size_t)HO * HP * 2);
  unsigned short* Ur_bf   = (unsigned short*)alloc((size_t)HO * HP * 2);
  unsigned short* Wz_bf   = (unsigned short*)alloc((size_t)HO * 960 * 2);
  unsigned short* Wh_bf   = (unsigned short*)alloc((size_t)HO * 960 * 2);
  unsigned short* Ww_bf   = (unsigned short*)alloc((size_t)HO * 544 * 2);
  unsigned short* Uw_bf   = (unsigned short*)alloc((size_t)HO * 1024 * 2);
  unsigned short* Wo_bf   = (unsigned short*)alloc((size_t)VP * HP * 2);
  unsigned short* Us_bf   = (unsigned short*)alloc((size_t)16 * HP * 2);
  float*          srcxr   = (float*)alloc((size_t)B_ * EPT * HO * 4);
  float*          h_buf   = (float*)alloc((size_t)B_ * EPT * HO * 4);
  unsigned short* hnei    = (unsigned short*)alloc((size_t)2 * B_ * HP * 2);
  float*          r_buf   = (float*)alloc((size_t)2 * B_ * HO * 4);
  float*          sum_h   = (float*)alloc((size_t)B_ * HO * 4);
  unsigned short* Az      = (unsigned short*)alloc((size_t)B_ * 960 * 2);
  unsigned short* Ah      = (unsigned short*)alloc((size_t)B_ * 960 * 2);
  float*          z_buf   = (float*)alloc((size_t)B_ * HO * 4);
  float*          pre_buf = (float*)alloc((size_t)B_ * HO * 4);
  unsigned short* pred_buf= (unsigned short*)alloc((size_t)16 * B_ * 544 * 2);
  unsigned short* stop_buf= (unsigned short*)alloc((size_t)31 * B_ * 1024 * 2);
  unsigned short* pred_hid= (unsigned short*)alloc((size_t)16 * B_ * HP * 2);
  float*          scores  = (float*)alloc((size_t)16 * B_ * VP * 4);
  unsigned short* stop_hid= (unsigned short*)alloc((size_t)31 * B_ * HP * 2);
  float*          s_buf   = (float*)alloc((size_t)31 * B_ * 16 * 4);
  int*            tgt     = (int*)alloc((size_t)16 * B_ * 4);
  float*          pred_pt = (float*)alloc((size_t)2048 * 2 * 4);
  float*          stop_pt = (float*)alloc((size_t)124 * 2 * 4);

  auto gsz = [](long long n) { return dim3((unsigned)((n + 255) / 256)); };
  auto pack = [&](const float* src, int sld, int sc0, unsigned short* dst, int dld, int dc0,
                  int rows, int cols) {
    k_pack<<<gsz((long long)rows * cols), 256, 0, stream>>>(src, sld, sc0, dst, dld, dc0, rows, cols);
  };
  auto gemm = [&](const unsigned short* A, int lda, int amode,
                  const unsigned short* Wt, int ldw, const float* bias,
                  const float* addend, int ld_add, int add_e0,
                  float* C, int ldc, unsigned short* Cbf, int ldcbf,
                  int M, int Ncols, int Ntiles, int Kpad, int act) {
    dim3 g(M / 16, (Ntiles + 3) / 4);
    gemm_bf16_wmma<<<g, 128, 0, stream>>>(A, lda, amode, Wt, ldw, bias, addend, ld_add, add_e0,
                                          C, ldc, Cbf, ldcbf, Ncols, Ntiles, Kpad, act);
  };

  // ---- setup: zero pads, pack weights to padded bf16, embed, tv, targets ----
  hipMemsetAsync(Wr_bf, 0, (size_t)HO * HP * 2, stream);
  hipMemsetAsync(Ur_bf, 0, (size_t)HO * HP * 2, stream);
  hipMemsetAsync(Wz_bf, 0, (size_t)HO * 960 * 2, stream);
  hipMemsetAsync(Wh_bf, 0, (size_t)HO * 960 * 2, stream);
  hipMemsetAsync(Ww_bf, 0, (size_t)HO * 544 * 2, stream);
  hipMemsetAsync(Uw_bf, 0, (size_t)HO * 1024 * 2, stream);
  hipMemsetAsync(Wo_bf, 0, (size_t)VP * HP * 2, stream);
  hipMemsetAsync(Us_bf, 0, (size_t)16 * HP * 2, stream);
  hipMemsetAsync(h_buf, 0, (size_t)B_ * EPT * HO * 4, stream);

  pack(W_r,   H_, 0,   Wr_bf, HP,  0,   H_, H_);
  pack(U_r_w, H_, 0,   Ur_bf, HP,  0,   H_, H_);
  pack(W_z_w, 900, 0,   Wz_bf, 960, 0,   H_, H_);
  pack(W_z_w, 900, 450, Wz_bf, 960, 480, H_, H_);
  pack(W_h_w, 900, 0,   Wh_bf, 960, 0,   H_, H_);
  pack(W_h_w, 900, 450, Wh_bf, 960, 480, H_, H_);
  pack(W_w,   506, 0,   Ww_bf, 544, 0,   H_, H_);
  pack(W_w,   506, 450, Ww_bf, 544, 480, H_, L_);
  pack(U_w,   956, 0,   Uw_bf, 1024, 0,   H_, H_);
  pack(U_w,   956, 450, Uw_bf, 1024, 480, H_, H_);
  pack(U_w,   956, 900, Uw_bf, 1024, 960, H_, L_);
  pack(W_o_w, H_, 0,   Wo_bf, HP, 0, V_, H_);
  pack(U_s_w, H_, 0,   Us_bf, HP, 0, 1,  H_);

  k_embed<<<gsz((long long)B_ * NN * HP), 256, 0, stream>>>(wid, emb, x_bf);
  k_tv<<<gsz((long long)B_ * LP), 256, 0, stream>>>(tree, tv_bf);
  k_tgt<<<gsz(16 * B_), 256, 0, stream>>>(wid, tgt);
  k_pred_row0<<<gsz((long long)B_ * 544), 256, 0, stream>>>(tv_bf, pred_buf);

  // src_x_r = x[g_src] @ W_r^T   (gather via amode=1)
  gemm(x_bf, HP, 1, Wr_bf, HP, nullptr, nullptr, 0, -1,
       srcxr, HO, nullptr, 0, B_ * EPT, H_, HO / 16, HP, 0);

  // ---- sequential DFS over 30 edges, batched over all 1024 trees ----
  int predk = 0;
  for (int t = 0; t < 30; ++t) {
    int e0 = SCHED_E[t], p = SCHED_P[t];
    int d = EDGE_D[e0], p0 = EDGE_P0[e0], p1 = EDGE_P1[e0];
    k_step_gather<<<gsz((long long)B_ * HP), 256, 0, stream>>>(h_buf, x_bf, hnei, sum_h, Az, Ah,
                                                               e0, d, p0, p1);
    if (d > 0) {
      // r = sigmoid(src_x_r[eid] + h_nei @ U_r^T + U_r_b)
      gemm(hnei, HP, 0, Ur_bf, HP, U_r_b, srcxr, HO, e0,
           r_buf, HO, nullptr, 0, d * B_, H_, HO / 16, HP, 1);
      k_sumg<<<gsz((long long)B_ * HP), 256, 0, stream>>>(h_buf, r_buf, Ah, d, p0, p1);
    }
    gemm(Az, 960, 0, Wz_bf, 960, W_z_b, nullptr, 0, -1,
         z_buf, HO, nullptr, 0, B_, H_, HO / 16, 960, 1);            // z = sigmoid
    gemm(Ah, 960, 0, Wh_bf, 960, W_h_b, nullptr, 0, -1,
         pre_buf, HO, nullptr, 0, B_, H_, HO / 16, 960, 2);          // pre = tanh
    int pk = -1;
    if (p == 0) { predk += 1; pk = predk; }
    k_step_final<<<gsz((long long)B_ * 1024), 256, 0, stream>>>(z_buf, pre_buf, sum_h, h_buf,
                                                                x_bf, tv_bf, stop_buf, pred_buf,
                                                                e0, t, pk);
  }
  k_root_stop<<<gsz((long long)B_ * 1024), 256, 0, stream>>>(h_buf, x_bf, tv_bf, stop_buf);

  // ---- heads ----
  gemm(pred_buf, 544, 0, Ww_bf, 544, W_b, nullptr, 0, -1,
       nullptr, 0, pred_hid, HP, 16 * B_, H_, HP / 16, 544, 3);      // relu
  gemm(pred_hid, HP, 0, Wo_bf, HP, W_o_b, nullptr, 0, -1,
       scores, VP, nullptr, 0, 16 * B_, V_, VP / 16, HP, 0);
  gemm(stop_buf, 1024, 0, Uw_bf, 1024, U_b, nullptr, 0, -1,
       nullptr, 0, stop_hid, HP, 31 * B_, H_, HP / 16, 1024, 3);     // relu
  gemm(stop_hid, HP, 0, Us_bf, HP, U_s_b, nullptr, 0, -1,
       s_buf, 16, nullptr, 0, 31 * B_, 1, 1, HP, 0);

  // ---- deterministic reductions -> 4 scalars ----
  k_pred_reduce<<<2048, 256, 0, stream>>>(scores, tgt, pred_pt);
  k_stop_reduce<<<124, 256, 0, stream>>>(s_buf, stop_pt);
  k_final<<<1, 32, 0, stream>>>(pred_pt, stop_pt, out);
}
